// Net_36696200577527
// MI455X (gfx1250) — matrix operand order repaired and checked by reading, not compiled, not run on hardware
//
#include <hip/hip_runtime.h>
#include <cmath>
#include <cstdint>

// ---------------- problem constants (from reference) ----------------
#define NN      50000
#define EE      160000
#define BBATCH  32
#define CC1     25000      // N/2
#define CC2     12500      // C1/2
#define INC     32
#define H1C     32
#define H2C     64
#define HIDC    25
#define FCC     128
#define OUTC    32
#define W2A_COLS (INC*H1C)   // 1024
#define W2B_COLS (H1C*H2C)   // 2048

typedef _Float16 half_t;
typedef __attribute__((ext_vector_type(16))) _Float16 v16h;
typedef __attribute__((ext_vector_type(8)))  _Float16 v8h;
typedef __attribute__((ext_vector_type(8)))  float    v8f;

// ---------------- small device helpers ----------------
__device__ __forceinline__ void atomicMaxF(float* addr, float v) {
  // IEEE ordering trick: positives ordered as int, negatives reverse-ordered as uint.
  if (v >= 0.0f) atomicMax((int*)addr, __float_as_int(v));
  else           atomicMin((unsigned int*)addr, __float_as_uint(v));
}
__device__ __forceinline__ float fix_empty(float v) {
  // segment_max empty (-inf) -> 0.0, per reference's where(isfinite(r), r, 0)
  return (v >= -3.0e38f) ? v : 0.0f;
}
__device__ __forceinline__ float elu1(float v) {
  return v > 0.0f ? v : (__expf(v) - 1.0f);
}

// A fragment (16x32 f16, row-major source row of 32 halves)
__device__ __forceinline__ v16h load_a_frag(const half_t* __restrict__ row, int lane) {
  const int kb = (lane < 16) ? 0 : 8;
  v8h lo = *(const v8h*)(row + kb);        // K = kb..kb+7
  v8h hi = *(const v8h*)(row + kb + 16);   // K = kb+16..kb+23
  return __builtin_shufflevector(lo, hi, 0,1,2,3,4,5,6,7,8,9,10,11,12,13,14,15);
}
// B fragment (32x16 f16) from pre-transposed weights w2T[col][k], k contiguous
__device__ __forceinline__ v16h load_b_frag(const half_t* __restrict__ colrow, int lane) {
  const int kb = (lane < 16) ? 0 : 8;
  v8h lo = *(const v8h*)(colrow + kb);        // K = kb..kb+7
  v8h hi = *(const v8h*)(colrow + kb + 16);   // K = kb+16..kb+23
  return __builtin_shufflevector(lo, hi, 0,1,2,3,4,5,6,7,8,9,10,11,12,13,14,15);
}

// ---------------- init fills ----------------
__global__ void k_fill_f32(float* p, int n, float v) {
  int i = blockIdx.x * blockDim.x + threadIdx.x;
  if (i < n) p[i] = v;
}
__global__ void k_fill_i32(int* p, int n, int v) {
  int i = blockIdx.x * blockDim.x + threadIdx.x;
  if (i < n) p[i] = v;
}

// ---------------- prep: transpose edge-MLP layer-2 weights to f16 [Ncol][32] ----------------
__global__ void k_prepw(const float* __restrict__ w2, half_t* __restrict__ w2T, int Ncol) {
  int t = blockIdx.x * blockDim.x + threadIdx.x;       // Ncol*32 threads
  if (t >= Ncol * 32) return;
  int c = t >> 5, k = t & 31;
  w2T[c * 32 + k] = (k < HIDC) ? (half_t)w2[k * Ncol + c] : (half_t)0.0f;
}

// ---------------- conv1 edge-MLP hidden: h1[E][32] = relu(ea@w1+b1), K-padded ----------------
__global__ void k_h1(const float* __restrict__ ea, const float* __restrict__ w1,
                     const float* __restrict__ b1, half_t* __restrict__ h1) {
  int t = blockIdx.x * blockDim.x + threadIdx.x;       // E*32
  if (t >= EE * 32) return;
  int e = t >> 5, k = t & 31;
  float v = 0.0f;
  if (k < HIDC) {
    v = fmaf(ea[e*2+0], w1[k], fmaf(ea[e*2+1], w1[HIDC + k], b1[k]));
    v = fmaxf(v, 0.0f);
  }
  h1[t] = (half_t)v;
}

// ---------------- conv1 fused: WMMA We-tile -> LDS -> per-edge matvec -> atomic max ----------------
__global__ void __launch_bounds__(256) k_conv1(
    const half_t* __restrict__ h1,     // [E][32]
    const half_t* __restrict__ w2T,    // [1024][32]
    const float*  __restrict__ b2,     // [1024]
    const float*  __restrict__ x,      // [N][32]
    const int*    __restrict__ src, const int* __restrict__ dst,
    float* __restrict__ msg)           // [N][32], init -inf
{
  extern __shared__ char smem[];
  float* We  = (float*)smem;                                  // 32*1024
  float* Xg  = (float*)(smem + 32*1024*4);                    // 32*32
  int*   sid = (int*)  (smem + 32*1024*4 + 32*32*4);          // 32
  int*   did = sid + 32;

  const int tid  = threadIdx.x;
  const int wave = tid >> 5;
  const int lane = tid & 31;
  const int e0   = blockIdx.x * 32;

  if (tid < 32) { sid[tid] = src[e0 + tid]; did[tid] = dst[e0 + tid]; }
  __syncthreads();
  #pragma unroll
  for (int j = 0; j < 4; ++j) {            // gather x[src] rows: 32 edges x 32 feats
    int li = tid * 4 + j;
    int e = li >> 5, i = li & 31;
    Xg[li] = x[(size_t)sid[e] * 32 + i];
  }

  // ---- WMMA phase: We[32][1024] = h1_tile @ w2 + b2 ----
  const int rt = wave & 1;                 // row tile (16 edges each)
  const int m  = lane & 15;
  const half_t* arow = h1 + (size_t)(e0 + rt*16 + m) * 32;
  v16h afrag = load_a_frag(arow, lane);

  for (int t = 0; t < 16; ++t) {
    const int ct = (wave >> 1) + 4 * t;    // 0..63
    const int colbase = ct * 16;
    if (t < 15)
      __builtin_prefetch(w2T + (size_t)(colbase + 64 + m) * 32, 0, 1);
    v16h bfrag = load_b_frag(w2T + (size_t)(colbase + m) * 32, lane);
    float bias = b2[colbase + m];
    v8f c;
    #pragma unroll
    for (int r = 0; r < 8; ++r) c[r] = bias;
    v8f d = __builtin_amdgcn_wmma_f32_16x16x32_f16(false, afrag, false, bfrag,
                                                   (short)0, c, false, false);
    const int erow = rt*16 + ((lane >= 16) ? 8 : 0);
    #pragma unroll
    for (int r = 0; r < 8; ++r)
      We[(erow + r) * 1024 + colbase + m] = d[r];
  }
  __syncthreads();

  // ---- contraction: msg[e,o] = sum_i Xg[e,i] * We[e, i*32+o] ----
  const int e  = tid >> 3;                 // 0..31
  const int ob = (tid & 7) * 4;            // 0..28
  float xr[32];
  #pragma unroll
  for (int i = 0; i < 32; ++i) xr[i] = Xg[e*32 + i];
  float acc[4] = {0.f, 0.f, 0.f, 0.f};
  #pragma unroll
  for (int i = 0; i < 32; ++i) {
    float xi = xr[i];
    #pragma unroll
    for (int j = 0; j < 4; ++j)
      acc[j] = fmaf(xi, We[e*1024 + i*32 + ob + j], acc[j]);
  }
  const int dn = did[e];
  #pragma unroll
  for (int j = 0; j < 4; ++j) atomicMaxF(&msg[(size_t)dn * 32 + ob + j], acc[j]);
}

// ---------------- post-conv: x_out = elu(fix(msg) + x_in@root + bias) ----------------
__global__ void k_post(const float* __restrict__ msg, const float* __restrict__ xin,
                       const float* __restrict__ root, const float* __restrict__ bias,
                       float* __restrict__ xout, int n, int inC, int outC) {
  int t = blockIdx.x * blockDim.x + threadIdx.x;
  if (t >= n * outC) return;
  int nd = t / outC, o = t % outC;
  float s = bias[o];
  const float* xr = xin + (size_t)nd * inC;
  for (int i = 0; i < inC; ++i) s = fmaf(xr[i], root[i * outC + o], s);
  xout[t] = elu1(fix_empty(msg[t]) + s);
}

// ---------------- max_pool(cluster1): x max, pos sum, cnt, batch max ----------------
__global__ void k_pool1(const float* __restrict__ x1, const float* __restrict__ pos,
                        const int* __restrict__ batch, const int* __restrict__ cl,
                        float* __restrict__ x2, float* __restrict__ pos2,
                        float* __restrict__ cnt, int* __restrict__ batch2) {
  int t = blockIdx.x * blockDim.x + threadIdx.x;
  if (t >= NN * 32) return;
  int nd = t >> 5, o = t & 31;
  int c = cl[nd];
  atomicMaxF(&x2[(size_t)c * 32 + o], x1[t]);
  if (o < 2)  atomicAdd(&pos2[(size_t)c * 2 + o], pos[(size_t)nd * 2 + o]);
  if (o == 0) { atomicAdd(&cnt[c], 1.0f); atomicMax(&batch2[c], batch[nd]); }
}
__global__ void k_pool1fin(float* __restrict__ x2, float* __restrict__ pos2,
                           const float* __restrict__ cnt) {
  int t = blockIdx.x * blockDim.x + threadIdx.x;
  if (t >= CC1 * 32) return;
  int c = t >> 5, o = t & 31;
  x2[t] = fix_empty(x2[t]);
  if (o < 2) pos2[c * 2 + o] /= fmaxf(cnt[c], 1.0f);
}

// ---------------- pooled-edge geometry: relabel + global max|cart| ----------------
__global__ void k_geom(const int* __restrict__ src, const int* __restrict__ dst,
                       const int* __restrict__ cl, const float* __restrict__ pos2,
                       int* __restrict__ src2, int* __restrict__ dst2,
                       unsigned* __restrict__ mx) {
  int e = blockIdx.x * blockDim.x + threadIdx.x;
  if (e >= EE) return;
  int s2 = cl[src[e]], d2 = cl[dst[e]];
  src2[e] = s2; dst2[e] = d2;
  float c0 = pos2[d2*2+0] - pos2[s2*2+0];
  float c1 = pos2[d2*2+1] - pos2[s2*2+1];
  atomicMax(mx, __float_as_uint(fabsf(c0)));
  atomicMax(mx, __float_as_uint(fabsf(c1)));
}

// ---------------- conv2 edge-MLP hidden from Cartesian attrs ----------------
__global__ void k_h2(const int* __restrict__ src2, const int* __restrict__ dst2,
                     const float* __restrict__ pos2, const unsigned* __restrict__ mx,
                     const float* __restrict__ w1, const float* __restrict__ b1,
                     half_t* __restrict__ h2) {
  int t = blockIdx.x * blockDim.x + threadIdx.x;
  if (t >= EE * 32) return;
  int e = t >> 5, k = t & 31;
  float inv = 1.0f / (2.0f * __uint_as_float(*mx));
  int s2 = src2[e], d2 = dst2[e];
  float a0 = (pos2[d2*2+0] - pos2[s2*2+0]) * inv + 0.5f;
  float a1 = (pos2[d2*2+1] - pos2[s2*2+1]) * inv + 0.5f;
  float v = 0.0f;
  if (k < HIDC) {
    v = fmaf(a0, w1[k], fmaf(a1, w1[HIDC + k], b1[k]));
    v = fmaxf(v, 0.0f);
  }
  h2[t] = (half_t)v;
}

// ---------------- conv2 fused (16-edge tiles, 2048 cols, self-loops masked) ----------------
__global__ void __launch_bounds__(256) k_conv2(
    const half_t* __restrict__ h2,     // [E][32]
    const half_t* __restrict__ w2T,    // [2048][32]
    const float*  __restrict__ b2,     // [2048]
    const float*  __restrict__ x2,     // [C1][32]
    const int*    __restrict__ src2, const int* __restrict__ dst2,
    float* __restrict__ msg)           // [C1][64], init -inf
{
  extern __shared__ char smem[];
  float* We  = (float*)smem;                                  // 16*2048
  float* Xg  = (float*)(smem + 16*2048*4);                    // 16*32
  int*   sid = (int*)  (smem + 16*2048*4 + 16*32*4);          // 16
  int*   did = sid + 16;

  const int tid  = threadIdx.x;
  const int wave = tid >> 5;
  const int lane = tid & 31;
  const int e0   = blockIdx.x * 16;

  if (tid < 16) { sid[tid] = src2[e0 + tid]; did[tid] = dst2[e0 + tid]; }
  __syncthreads();
  #pragma unroll
  for (int j = 0; j < 2; ++j) {            // gather x2[src2] rows: 16 x 32
    int li = tid * 2 + j;
    int e = li >> 5, i = li & 31;
    Xg[li] = x2[(size_t)sid[e] * 32 + i];
  }

  const int m = lane & 15;
  const half_t* arow = h2 + (size_t)(e0 + m) * 32;
  v16h afrag = load_a_frag(arow, lane);

  for (int t = 0; t < 16; ++t) {
    const int ct = wave + 8 * t;           // 0..127
    const int colbase = ct * 16;
    if (t < 15)
      __builtin_prefetch(w2T + (size_t)(colbase + 128 + m) * 32, 0, 1);
    v16h bfrag = load_b_frag(w2T + (size_t)(colbase + m) * 32, lane);
    float bias = b2[colbase + m];
    v8f c;
    #pragma unroll
    for (int r = 0; r < 8; ++r) c[r] = bias;
    v8f d = __builtin_amdgcn_wmma_f32_16x16x32_f16(false, afrag, false, bfrag,
                                                   (short)0, c, false, false);
    const int erow = (lane >= 16) ? 8 : 0;
    #pragma unroll
    for (int r = 0; r < 8; ++r)
      We[(erow + r) * 2048 + colbase + m] = d[r];
  }
  __syncthreads();

  const int e  = tid >> 4;                 // 0..15
  const int ob = (tid & 15) * 4;           // 0..60
  float xr[32];
  #pragma unroll
  for (int i = 0; i < 32; ++i) xr[i] = Xg[e*32 + i];
  float acc[4] = {0.f, 0.f, 0.f, 0.f};
  #pragma unroll
  for (int i = 0; i < 32; ++i) {
    float xi = xr[i];
    #pragma unroll
    for (int j = 0; j < 4; ++j)
      acc[j] = fmaf(xi, We[e*2048 + i*64 + ob + j], acc[j]);
  }
  if (sid[e] != did[e]) {                  // remove_self_loops: skip masked edges
    const int dn = did[e];
    #pragma unroll
    for (int j = 0; j < 4; ++j) atomicMaxF(&msg[(size_t)dn * 64 + ob + j], acc[j]);
  }
}

// ---------------- max_pool_x(cluster2) ----------------
__global__ void k_pool2(const float* __restrict__ x3, const int* __restrict__ cl2,
                        const int* __restrict__ batch2,
                        float* __restrict__ x4, int* __restrict__ batch3) {
  int t = blockIdx.x * blockDim.x + threadIdx.x;
  if (t >= CC1 * 64) return;
  int nd = t >> 6, o = t & 63;
  int c = cl2[nd];
  atomicMaxF(&x4[(size_t)c * 64 + o], x3[t]);
  if (o == 0) atomicMax(&batch3[c], batch2[nd]);
}
__global__ void k_pool2fin(float* __restrict__ x4, const int* __restrict__ batch3,
                           float* __restrict__ gsum, float* __restrict__ gcnt) {
  int t = blockIdx.x * blockDim.x + threadIdx.x;
  if (t >= CC2 * 64) return;
  int c = t >> 6, o = t & 63;
  float v = fix_empty(x4[t]);
  x4[t] = v;
  int b = batch3[c];
  if (b >= 0) {
    atomicAdd(&gsum[(size_t)b * 64 + o], v);
    if (o == 0) atomicAdd(&gcnt[b], 1.0f);
  }
}

// ---------------- head: mean pool -> fc1+elu -> fc2 -> log_softmax ----------------
__global__ void __launch_bounds__(256) k_head(
    const float* __restrict__ gsum, const float* __restrict__ gcnt,
    const float* __restrict__ fc1w, const float* __restrict__ fc1b,
    const float* __restrict__ fc2w, const float* __restrict__ fc2b,
    float* __restrict__ out) {
  __shared__ float g[BBATCH * H2C];     // 32x64
  __shared__ float h[BBATCH * FCC];     // 32x128
  __shared__ float lg[BBATCH * OUTC];   // 32x32
  int t = threadIdx.x;
  for (int li = t; li < BBATCH * H2C; li += 256) {
    int r = li >> 6;
    g[li] = gsum[li] / fmaxf(gcnt[r], 1.0f);
  }
  __syncthreads();
  for (int li = t; li < BBATCH * FCC; li += 256) {
    int r = li >> 7, c = li & 127;
    float s = fc1b[c];
    for (int i = 0; i < H2C; ++i) s = fmaf(g[r * H2C + i], fc1w[i * FCC + c], s);
    h[li] = elu1(s);
  }
  __syncthreads();
  for (int li = t; li < BBATCH * OUTC; li += 256) {
    int r = li >> 5, c = li & 31;
    float s = fc2b[c];
    for (int i = 0; i < FCC; ++i) s = fmaf(h[r * FCC + i], fc2w[i * OUTC + c], s);
    lg[li] = s;
  }
  __syncthreads();
  if (t < BBATCH) {
    float m = -3.4e38f;
    for (int c = 0; c < OUTC; ++c) m = fmaxf(m, lg[t * OUTC + c]);
    float s = 0.0f;
    for (int c = 0; c < OUTC; ++c) s += __expf(lg[t * OUTC + c] - m);
    float ls = __logf(s);
    for (int c = 0; c < OUTC; ++c) out[t * OUTC + c] = lg[t * OUTC + c] - m - ls;
  }
}

// ---------------- host launcher ----------------
extern "C" void kernel_launch(void* const* d_in, const int* in_sizes, int n_in,
                              void* d_out, int out_size, void* d_ws, size_t ws_size,
                              hipStream_t stream) {
  (void)in_sizes; (void)n_in; (void)out_size; (void)ws_size;
  const float* x     = (const float*)d_in[0];
  const float* pos   = (const float*)d_in[1];
  const float* ea    = (const float*)d_in[2];
  const int*   eidx  = (const int*)d_in[3];
  const int*   batch = (const int*)d_in[4];
  const int*   cl1   = (const int*)d_in[5];
  const int*   cl2   = (const int*)d_in[6];
  const float* n1w1  = (const float*)d_in[7];
  const float* n1b1  = (const float*)d_in[8];
  const float* n1w2  = (const float*)d_in[9];
  const float* n1b2  = (const float*)d_in[10];
  const float* root1 = (const float*)d_in[11];
  const float* bias1 = (const float*)d_in[12];
  const float* n2w1  = (const float*)d_in[13];
  const float* n2b1  = (const float*)d_in[14];
  const float* n2w2  = (const float*)d_in[15];
  const float* n2b2  = (const float*)d_in[16];
  const float* root2 = (const float*)d_in[17];
  const float* bias2 = (const float*)d_in[18];
  const float* fc1w  = (const float*)d_in[19];
  const float* fc1b  = (const float*)d_in[20];
  const float* fc2w  = (const float*)d_in[21];
  const float* fc2b  = (const float*)d_in[22];
  const int* src = eidx;
  const int* dst = eidx + EE;
  float* out = (float*)d_out;

  // workspace carve (256-byte aligned)
  char* w = (char*)d_ws;
  auto carve = [&](size_t bytes) -> char* {
    char* p = w; w += (bytes + 255) & ~(size_t)255; return p;
  };
  half_t*   h1     = (half_t*)carve((size_t)EE * 32 * sizeof(half_t));
  half_t*   h2     = (half_t*)carve((size_t)EE * 32 * sizeof(half_t));
  half_t*   w2aT   = (half_t*)carve((size_t)W2A_COLS * 32 * sizeof(half_t));
  half_t*   w2bT   = (half_t*)carve((size_t)W2B_COLS * 32 * sizeof(half_t));
  float*    msg1   = (float*)carve((size_t)NN * H1C * sizeof(float));
  float*    x1     = (float*)carve((size_t)NN * H1C * sizeof(float));
  float*    x2     = (float*)carve((size_t)CC1 * H1C * sizeof(float));
  float*    pos2   = (float*)carve((size_t)CC1 * 2 * sizeof(float));
  float*    cnt    = (float*)carve((size_t)CC1 * sizeof(float));
  int*      batch2 = (int*)carve((size_t)CC1 * sizeof(int));
  int*      src2   = (int*)carve((size_t)EE * sizeof(int));
  int*      dst2   = (int*)carve((size_t)EE * sizeof(int));
  float*    msg2   = (float*)carve((size_t)CC1 * H2C * sizeof(float));
  float*    x3     = (float*)carve((size_t)CC1 * H2C * sizeof(float));
  float*    x4     = (float*)carve((size_t)CC2 * H2C * sizeof(float));
  int*      batch3 = (int*)carve((size_t)CC2 * sizeof(int));
  float*    gsum   = (float*)carve((size_t)BBATCH * H2C * sizeof(float));
  float*    gcnt   = (float*)carve((size_t)BBATCH * sizeof(float));
  unsigned* mx     = (unsigned*)carve(sizeof(unsigned));

  auto cdiv = [](int a, int b) { return (a + b - 1) / b; };
  const float NEGINF = -INFINITY;

  // init accumulators
  k_fill_f32<<<cdiv(NN*H1C,256),256,0,stream>>>(msg1, NN*H1C, NEGINF);
  k_fill_f32<<<cdiv(CC1*H1C,256),256,0,stream>>>(x2, CC1*H1C, NEGINF);
  k_fill_f32<<<cdiv(CC1*H2C,256),256,0,stream>>>(msg2, CC1*H2C, NEGINF);
  k_fill_f32<<<cdiv(CC2*H2C,256),256,0,stream>>>(x4, CC2*H2C, NEGINF);
  k_fill_f32<<<cdiv(CC1*2,256),256,0,stream>>>(pos2, CC1*2, 0.0f);
  k_fill_f32<<<cdiv(CC1,256),256,0,stream>>>(cnt, CC1, 0.0f);
  k_fill_f32<<<cdiv(BBATCH*H2C,256),256,0,stream>>>(gsum, BBATCH*H2C, 0.0f);
  k_fill_f32<<<1,256,0,stream>>>(gcnt, BBATCH, 0.0f);
  k_fill_i32<<<cdiv(CC1,256),256,0,stream>>>(batch2, CC1, (int)0x80000000);
  k_fill_i32<<<cdiv(CC2,256),256,0,stream>>>(batch3, CC2, (int)0x80000000);
  k_fill_i32<<<1,32,0,stream>>>((int*)mx, 1, 0);

  // prep weights + conv1 hidden
  k_prepw<<<cdiv(W2A_COLS*32,256),256,0,stream>>>(n1w2, w2aT, W2A_COLS);
  k_prepw<<<cdiv(W2B_COLS*32,256),256,0,stream>>>(n2w2, w2bT, W2B_COLS);
  k_h1<<<cdiv(EE*32,256),256,0,stream>>>(ea, n1w1, n1b1, h1);

  // conv1 (WMMA fused) + elu/root
  size_t sm1 = 32*1024*4 + 32*32*4 + 64*4;
  k_conv1<<<EE/32, 256, sm1, stream>>>(h1, w2aT, n1b2, x, src, dst, msg1);
  k_post<<<cdiv(NN*H1C,256),256,0,stream>>>(msg1, x, root1, bias1, x1, NN, INC, H1C);

  // max_pool(cluster1)
  k_pool1<<<cdiv(NN*32,256),256,0,stream>>>(x1, pos, batch, cl1, x2, pos2, cnt, batch2);
  k_pool1fin<<<cdiv(CC1*32,256),256,0,stream>>>(x2, pos2, cnt);

  // pooled-graph Cartesian attrs -> conv2 hidden
  k_geom<<<cdiv(EE,256),256,0,stream>>>(src, dst, cl1, pos2, src2, dst2, mx);
  k_h2<<<cdiv(EE*32,256),256,0,stream>>>(src2, dst2, pos2, mx, n2w1, n2b1, h2);

  // conv2 (WMMA fused) + elu/root
  size_t sm2 = 16*2048*4 + 16*32*4 + 32*4;
  k_conv2<<<EE/16, 256, sm2, stream>>>(h2, w2bT, n2b2, x2, src2, dst2, msg2);
  k_post<<<cdiv(CC1*H2C,256),256,0,stream>>>(msg2, x2, root2, bias2, x3, CC1, H1C, H2C);

  // max_pool_x(cluster2) -> global mean pool
  k_pool2<<<cdiv(CC1*64,256),256,0,stream>>>(x3, cl2, batch2, x4, batch3);
  k_pool2fin<<<cdiv(CC2*64,256),256,0,stream>>>(x4, batch3, gsum, gcnt);

  // FC head + log_softmax
  k_head<<<1, 256, 0, stream>>>(gsum, gcnt, fc1w, fc1b, fc2w, fc2b, out);
}
